// RcnnHead_39187281609222
// MI455X (gfx1250) — matrix operand order repaired and checked by reading, not compiled, not run on hardware
//
#include <hip/hip_runtime.h>
#include <hip/hip_bf16.h>
#include <stdint.h>

// ---------------------------------------------------------------------------
// Types for CDNA5 WMMA (wave32, v_wmma_f32_16x16x32_bf16)
// ---------------------------------------------------------------------------
typedef __attribute__((ext_vector_type(16))) __bf16 v16bf;
typedef __attribute__((ext_vector_type(8)))  __bf16 v8bf;
typedef __attribute__((ext_vector_type(8)))  float  v8f;
typedef __attribute__((ext_vector_type(4)))  float  f4v;
typedef __attribute__((ext_vector_type(4)))  unsigned short us4;
typedef __attribute__((ext_vector_type(4)))  unsigned int   u4v;
typedef __attribute__((ext_vector_type(4)))  int            i4v;

#define B_IMG 4
#define N_PROP 1024
#define C_HID 1024
#define FEAT_DIM 12544
#define NUM_CLASSES 21
#define IMG_WH 800.0f

#if __has_builtin(__builtin_amdgcn_global_load_async_to_lds_b128)
#define HAS_ASYNC_LDS 1
#else
#define HAS_ASYNC_LDS 0
#endif

typedef __attribute__((address_space(1))) i4v* g_i4p;
typedef __attribute__((address_space(3))) i4v* l_i4p;

__device__ __forceinline__ unsigned short f2bf(float f) {
    union { float f; uint32_t u; } v; v.f = f;
    uint32_t u = v.u;
    uint32_t r = u + 0x7FFFu + ((u >> 16) & 1u);   // round-to-nearest-even
    return (unsigned short)(r >> 16);
}

// 16-byte global -> LDS copy: async (ASYNCcnt) when available, sync fallback.
__device__ __forceinline__ void copy16_g2l(unsigned short* lds, const unsigned short* g) {
#if HAS_ASYNC_LDS
    __builtin_amdgcn_global_load_async_to_lds_b128(
        (g_i4p)g, (l_i4p)lds, /*offset=*/0, /*cpol=*/0);
#else
    *(u4v*)lds = *(const u4v*)g;
#endif
}

__device__ __forceinline__ void wait_async_copies() {
#if HAS_ASYNC_LDS && __has_builtin(__builtin_amdgcn_s_wait_asynccnt)
    __builtin_amdgcn_s_wait_asynccnt(0);
#endif
}

// ---------------------------------------------------------------------------
// All-bf16 tiled GEMM + bias + ReLU on v_wmma_f32_16x16x32_bf16.
//   C[M,N] = relu(A[M,K] * Bt[N,K]^T + bias[N])        (Bt is pre-transposed)
// Block tile 128x128, BK=32, 256 threads = 8 waves (2x4); each wave owns a
// 64x32 slab = 4x2 WMMA tiles. Double-buffered LDS; global->LDS staging uses
// GLOBAL_LOAD_ASYNC_TO_LDS_B128 when the toolchain exposes it.
// ---------------------------------------------------------------------------
template <bool OUT_BF16>
__global__ __launch_bounds__(256)
void gemm_bf16_wmma(const unsigned short* __restrict__ A,
                    const unsigned short* __restrict__ Bt,
                    const float* __restrict__ bias, void* __restrict__ Out,
                    int M, int N, int K)
{
    constexpr int BM = 128, BN = 128, BK = 32, LPAD = 8, STR = BK + LPAD; // 80B rows, 16B-aligned
    __shared__ unsigned short lA[2][BM * STR];
    __shared__ unsigned short lB[2][BN * STR];

    const int tid  = threadIdx.x;
    const int lane = tid & 31;
    const int wave = tid >> 5;
    const int wm   = wave >> 2;        // 0..1  (64-row slab)
    const int wn   = wave & 3;         // 0..3  (32-col slab)
    const int mBlock = blockIdx.y * BM;
    const int nBlock = blockIdx.x * BN;

    v8f acc[4][2] = {};

    const int nK = K / BK;

    // Tile staging: 128 rows x 4 chunks of 16B per matrix; 256 threads -> 2 chunks each.
    auto stage = [&](int buf, int kBase) {
#pragma unroll
        for (int p = 0; p < 2; ++p) {
            const int c   = tid + p * 256;      // 0..511
            const int row = c >> 2;             // 0..127
            const int h   = (c & 3) * 8;        // half-offset {0,8,16,24}
            copy16_g2l(&lA[buf][row * STR + h], &A [(size_t)(mBlock + row) * K + kBase + h]);
            copy16_g2l(&lB[buf][row * STR + h], &Bt[(size_t)(nBlock + row) * K + kBase + h]);
        }
    };

    stage(0, 0);

    const int mWave = wm * 64, nWave = wn * 32;
    const int lrow  = lane & 15;
    // A frag layout (16-bit 16x32): lanes 0-15 -> K {0..7,16..23}; lanes 16-31 -> K {8..15,24..31}
    const int akh   = (lane & 16) ? 8 : 0;
    // B frag layout (16-bit 32x16): lanes 0-15 -> K {0..7,16..23}; lanes 16-31 -> K {8..15,24..31}
    const int bkh   = (lane & 16) ? 8 : 0;

    for (int kt = 0; kt < nK; ++kt) {
        const int buf = kt & 1;

        wait_async_copies();     // this wave's staged copies for `buf` are done
        __syncthreads();         // everyone's copies visible; prior buffer reads finished

        if (kt + 1 < nK) stage(buf ^ 1, (kt + 1) * BK);

        // ---- fragment loads + 8 WMMAs on current buffer ----
        v16bf bfrag[2];
#pragma unroll
        for (int nt = 0; nt < 2; ++nt) {
            const unsigned short* bp = &lB[buf][(nWave + nt * 16 + lrow) * STR + bkh];
            v8bf b0 = *(const v8bf*)(bp);
            v8bf b1 = *(const v8bf*)(bp + 16);
#pragma unroll
            for (int i = 0; i < 8; ++i) { bfrag[nt][i] = b0[i]; bfrag[nt][i + 8] = b1[i]; }
        }

#pragma unroll
        for (int mt = 0; mt < 4; ++mt) {
            const unsigned short* ap = &lA[buf][(mWave + mt * 16 + lrow) * STR + akh];
            v8bf a0 = *(const v8bf*)(ap);
            v8bf a1 = *(const v8bf*)(ap + 16);
            v16bf afrag;
#pragma unroll
            for (int i = 0; i < 8; ++i) { afrag[i] = a0[i]; afrag[i + 8] = a1[i]; }
#pragma unroll
            for (int nt = 0; nt < 2; ++nt) {
                acc[mt][nt] = __builtin_amdgcn_wmma_f32_16x16x32_bf16(
                    /*neg_a=*/false, afrag, /*neg_b=*/false, bfrag[nt],
                    /*c_mod=*/(short)0, acc[mt][nt], /*reuse_a=*/false, /*reuse_b=*/false);
            }
        }
    }

    // ---- epilogue: bias + ReLU ----
    // C layout: VGPR r -> lanes 0-15: (M=r, N=lane) ; lanes 16-31: (M=8+r, N=lane-16)
#pragma unroll
    for (int mt = 0; mt < 4; ++mt) {
#pragma unroll
        for (int nt = 0; nt < 2; ++nt) {
            const int n  = nBlock + wn * 32 + nt * 16 + (lane & 15);
            const float bs = bias[n];
            const int m0 = mBlock + wm * 64 + mt * 16 + ((lane & 16) ? 8 : 0);
#pragma unroll
            for (int r = 0; r < 8; ++r) {
                float v = acc[mt][nt][r] + bs;
                v = v > 0.0f ? v : 0.0f;
                const size_t off = (size_t)(m0 + r) * N + n;
                if (OUT_BF16) ((unsigned short*)Out)[off] = f2bf(v);
                else          ((float*)Out)[off] = v;
            }
        }
    }
}

// ---------------------------------------------------------------------------
// One-time fp32 -> bf16 conversion (row-major preserved), vectorized x4.
// ---------------------------------------------------------------------------
__global__ __launch_bounds__(256)
void cvt_f32_bf16(const float* __restrict__ in, unsigned short* __restrict__ out, long n4)
{
    const long i = (long)blockIdx.x * 256 + threadIdx.x;
    if (i >= n4) return;
    f4v v = ((const f4v*)in)[i];
    us4 o;
    o.x = f2bf(v.x); o.y = f2bf(v.y); o.z = f2bf(v.z); o.w = f2bf(v.w);
    ((us4*)out)[i] = o;
}

// ---------------------------------------------------------------------------
// One-time fp32[K][N] -> bf16[N][K] transpose+convert (32x32 LDS tiles).
// ---------------------------------------------------------------------------
__global__ __launch_bounds__(256)
void transpose_cvt_f32_bf16(const float* __restrict__ in, unsigned short* __restrict__ out,
                            int K, int N)
{
    __shared__ unsigned short t[32][33];
    const int kb = blockIdx.y * 32, nb = blockIdx.x * 32;
    const int tx = threadIdx.x, ty = threadIdx.y;   // block (32,8)
#pragma unroll
    for (int j = 0; j < 4; ++j) {
        const int row = ty + j * 8;
        t[row][tx] = f2bf(in[(size_t)(kb + row) * N + nb + tx]);
    }
    __syncthreads();
#pragma unroll
    for (int j = 0; j < 4; ++j) {
        const int row = ty + j * 8;
        out[(size_t)(nb + row) * K + kb + tx] = t[tx][row];
    }
}

// ---------------------------------------------------------------------------
// cls / box heads: per-row dot products (tiny: ~0.9 GFLOP total)
// ---------------------------------------------------------------------------
__global__ __launch_bounds__(128)
void heads_kernel(const float* __restrict__ X2,
                  const float* __restrict__ cls_w, const float* __restrict__ cls_b,
                  const float* __restrict__ box_w, const float* __restrict__ box_b,
                  float* __restrict__ clsL, float* __restrict__ boxD)
{
    __shared__ float x[C_HID];
    const int row = blockIdx.x;
    for (int k = threadIdx.x; k < C_HID; k += 128)
        x[k] = X2[(size_t)row * C_HID + k];
    __syncthreads();

    const int t = threadIdx.x;
    if (t < NUM_CLASSES) {
        float acc = cls_b[t];
        for (int k = 0; k < C_HID; ++k)
            acc = fmaf(x[k], cls_w[k * NUM_CLASSES + t], acc);
        clsL[row * NUM_CLASSES + t] = acc;
    } else if (t < NUM_CLASSES + NUM_CLASSES * 4) {
        const int u = t - NUM_CLASSES;
        float acc = box_b[u];
        for (int k = 0; k < C_HID; ++k)
            acc = fmaf(x[k], box_w[k * (NUM_CLASSES * 4) + u], acc);
        boxD[row * (NUM_CLASSES * 4) + u] = acc;
    }
}

// ---------------------------------------------------------------------------
// softmax + fg argmax + box decode + clip + validity; also fills rpn/rcnn props
// ---------------------------------------------------------------------------
__global__ void postproc_kernel(const float* __restrict__ clsL, const float* __restrict__ boxD,
                                const float* __restrict__ proposals,
                                float* __restrict__ bxv, float* __restrict__ scv,
                                int* __restrict__ labv, int* __restrict__ valv,
                                float* __restrict__ rpn, float* __restrict__ rcnn)
{
    const int idx = blockIdx.x * blockDim.x + threadIdx.x;
    if (idx >= B_IMG * N_PROP) return;

    const float* l = clsL + idx * NUM_CLASSES;
    float mx = l[0];
    for (int c = 1; c < NUM_CLASSES; ++c) mx = fmaxf(mx, l[c]);
    float e[NUM_CLASSES], s = 0.0f;
    for (int c = 0; c < NUM_CLASSES; ++c) { e[c] = __expf(l[c] - mx); s += e[c]; }
    const float inv = 1.0f / s;

    int best = 0; float bp = -1.0f;
    for (int c = 1; c < NUM_CLASSES; ++c) {
        float p = e[c] * inv;
        if (p > bp) { bp = p; best = c - 1; }     // argmax over fg classes
    }
    const int label = best + 1;
    const float sc = bp;

    const float* d = boxD + idx * (NUM_CLASSES * 4) + label * 4;
    const float d0 = d[0] * 0.1f, d1 = d[1] * 0.1f, d2 = d[2] * 0.2f, d3 = d[3] * 0.2f;

    const float* pr = proposals + idx * 4;
    const float w = pr[2] - pr[0], h = pr[3] - pr[1];
    const float cx = pr[0] + 0.5f * w, cy = pr[1] + 0.5f * h;
    const float pcx = d0 * w + cx, pcy = d1 * h + cy;
    const float pw = __expf(d2) * w, ph = __expf(d3) * h;
    float x0 = pcx - 0.5f * pw, y0 = pcy - 0.5f * ph;
    float x1 = pcx + 0.5f * pw, y1 = pcy + 0.5f * ph;
    x0 = fminf(fmaxf(x0, 0.0f), IMG_WH); y0 = fminf(fmaxf(y0, 0.0f), IMG_WH);
    x1 = fminf(fmaxf(x1, 0.0f), IMG_WH); y1 = fminf(fmaxf(y1, 0.0f), IMG_WH);

    const int valid = (x1 - x0 > 0.0f) && (y1 - y0 > 0.0f) && (sc > 0.1f);

    bxv[idx * 4 + 0] = x0; bxv[idx * 4 + 1] = y0;
    bxv[idx * 4 + 2] = x1; bxv[idx * 4 + 3] = y1;
    scv[idx] = sc; labv[idx] = label; valv[idx] = valid;

    rpn[idx * 6 + 0] = pr[0]; rpn[idx * 6 + 1] = pr[1];
    rpn[idx * 6 + 2] = pr[2]; rpn[idx * 6 + 3] = pr[3];
    rpn[idx * 6 + 4] = 0.0f;  rpn[idx * 6 + 5] = 0.0f;

    rcnn[idx * 6 + 0] = x0; rcnn[idx * 6 + 1] = y0;
    rcnn[idx * 6 + 2] = x1; rcnn[idx * 6 + 3] = y1;
    rcnn[idx * 6 + 4] = 0.0f; rcnn[idx * 6 + 5] = 0.0f;
}

// ---------------------------------------------------------------------------
// Per-image NMS: bitonic sort by score (desc) in LDS + sequential suppression.
// One 1024-thread block (32 waves) per batch image.
// ---------------------------------------------------------------------------
__global__ __launch_bounds__(1024)
void nms_kernel(const float* __restrict__ bxv, const float* __restrict__ scv,
                const int* __restrict__ labv, const int* __restrict__ valv,
                float* __restrict__ res, float* __restrict__ keep_out)
{
    __shared__ float sk[N_PROP];
    __shared__ int   si[N_PROP];
    __shared__ float sx0[N_PROP], sy0[N_PROP], sx1[N_PROP], sy1[N_PROP];
    __shared__ int   kp[N_PROP];

    const int b = blockIdx.x, tid = threadIdx.x;
    const int g = b * N_PROP + tid;

    sk[tid] = valv[g] ? scv[g] : -1.0f;
    si[tid] = tid;
    __syncthreads();

    // bitonic sort, descending by score
    for (int k = 2; k <= N_PROP; k <<= 1) {
        for (int j = k >> 1; j > 0; j >>= 1) {
            const int ixj = tid ^ j;
            if (ixj > tid) {
                const float a = sk[tid], c = sk[ixj];
                const bool descSeg = ((tid & k) == 0);
                if ((a < c) == descSeg) {
                    sk[tid] = c; sk[ixj] = a;
                    const int t = si[tid]; si[tid] = si[ixj]; si[ixj] = t;
                }
            }
            __syncthreads();
        }
    }

    // gather class-offset boxes in sorted order
    const int o  = si[tid];
    const int go = b * N_PROP + o;
    const float off = (float)labv[go] * (IMG_WH + 1.0f);
    sx0[tid] = bxv[go * 4 + 0] + off;
    sy0[tid] = bxv[go * 4 + 1] + off;
    sx1[tid] = bxv[go * 4 + 2] + off;
    sy1[tid] = bxv[go * 4 + 3] + off;
    kp[tid]  = valv[go];
    __syncthreads();

    const float ax0 = sx0[tid], ay0 = sy0[tid], ax1 = sx1[tid], ay1 = sy1[tid];
    const float areaA = (ax1 - ax0) * (ay1 - ay0);

    for (int i = 0; i < N_PROP - 1; ++i) {
        if (kp[i] && tid > i && kp[tid]) {
            const float lx = fmaxf(ax0, sx0[i]), ly = fmaxf(ay0, sy0[i]);
            const float rx = fminf(ax1, sx1[i]), ry = fminf(ay1, sy1[i]);
            const float iw = fmaxf(rx - lx, 0.0f), ih = fmaxf(ry - ly, 0.0f);
            const float inter = iw * ih;
            const float areaB = (sx1[i] - sx0[i]) * (sy1[i] - sy0[i]);
            const float iou = inter / fmaxf(areaA + areaB - inter, 1e-9f);
            if (iou > 0.5f) kp[tid] = 0;
        }
        __syncthreads();
    }

    // scatter back to original order; write keep + res
    const bool kept = (kp[tid] != 0);
    keep_out[go] = kept ? 1.0f : 0.0f;
    float* r = res + (size_t)go * 6;
    if (kept) {
        r[0] = bxv[go * 4 + 0]; r[1] = bxv[go * 4 + 1];
        r[2] = bxv[go * 4 + 2]; r[3] = bxv[go * 4 + 3];
        r[4] = (float)(labv[go] - 1); r[5] = scv[go];
    } else {
        r[0] = 0.0f; r[1] = 0.0f; r[2] = 0.0f; r[3] = 0.0f; r[4] = 0.0f; r[5] = 0.0f;
    }
}

// ---------------------------------------------------------------------------
// launch
// ---------------------------------------------------------------------------
extern "C" void kernel_launch(void* const* d_in, const int* in_sizes, int n_in,
                              void* d_out, int out_size, void* d_ws, size_t ws_size,
                              hipStream_t stream)
{
    (void)in_sizes; (void)n_in; (void)out_size; (void)ws_size;

    const float* feat      = (const float*)d_in[0];  // (4096, 12544)
    const float* proposals = (const float*)d_in[1];  // (4, 1024, 4)
    const float* fc6_w     = (const float*)d_in[2];  // (12544, 1024)
    const float* fc6_b     = (const float*)d_in[3];
    const float* fc7_w     = (const float*)d_in[4];  // (1024, 1024)
    const float* fc7_b     = (const float*)d_in[5];
    const float* cls_w     = (const float*)d_in[6];  // (1024, 21)
    const float* cls_b     = (const float*)d_in[7];
    const float* box_w     = (const float*)d_in[8];  // (1024, 84)
    const float* box_b     = (const float*)d_in[9];

    const int M = B_IMG * N_PROP;                    // 4096

    // workspace layout (bf16 staging + activations)
    char* ws = (char*)d_ws;
    unsigned short* featB = (unsigned short*)ws;                       // bf16 4096x12544
    unsigned short* w6t   = featB + (size_t)M * FEAT_DIM;              // bf16 1024x12544 (transposed)
    unsigned short* w7t   = w6t + (size_t)C_HID * FEAT_DIM;            // bf16 1024x1024  (transposed)
    unsigned short* X1    = w7t + (size_t)C_HID * C_HID;               // bf16 4096x1024
    float* X2   = (float*)(X1 + (size_t)M * C_HID);                    // f32 4096x1024
    float* clsL = X2 + (size_t)M * C_HID;
    float* boxD = clsL + (size_t)M * NUM_CLASSES;
    float* bxv  = boxD + (size_t)M * NUM_CLASSES * 4;
    float* scv  = bxv + (size_t)M * 4;
    int*   labv = (int*)(scv + M);
    int*   valv = labv + M;

    // output layout: res(4,1024,6) | keep(4,1024) | rpn(4,1024,6) | rcnn(4,1024,6)
    float* out  = (float*)d_out;
    float* res  = out;
    float* keep = res + (size_t)M * 6;
    float* rpn  = keep + M;
    float* rcnn = rpn + (size_t)M * 6;

    // ---- one-time bf16 staging ----
    {
        const long n4 = (long)M * FEAT_DIM / 4;
        cvt_f32_bf16<<<(unsigned)((n4 + 255) / 256), 256, 0, stream>>>(feat, featB, n4);
    }
    transpose_cvt_f32_bf16<<<dim3(C_HID / 32, FEAT_DIM / 32), dim3(32, 8), 0, stream>>>(
        fc6_w, w6t, FEAT_DIM, C_HID);
    transpose_cvt_f32_bf16<<<dim3(C_HID / 32, C_HID / 32), dim3(32, 8), 0, stream>>>(
        fc7_w, w7t, C_HID, C_HID);

    dim3 gGemm(C_HID / 128, M / 128);   // (8, 32)

    // fc6: bf16 x bf16^T, bf16 out
    gemm_bf16_wmma<true><<<gGemm, 256, 0, stream>>>(
        featB, w6t, fc6_b, (void*)X1, M, C_HID, FEAT_DIM);

    // fc7: bf16 x bf16^T, fp32 out
    gemm_bf16_wmma<false><<<gGemm, 256, 0, stream>>>(
        X1, w7t, fc7_b, (void*)X2, M, C_HID, C_HID);

    heads_kernel<<<M, 128, 0, stream>>>(X2, cls_w, cls_b, box_w, box_b, clsL, boxD);

    postproc_kernel<<<(M + 255) / 256, 256, 0, stream>>>(
        clsL, boxD, proposals, bxv, scv, labv, valv, rpn, rcnn);

    nms_kernel<<<B_IMG, N_PROP, 0, stream>>>(bxv, scv, labv, valv, res, keep);
}
